// MoEMLP_17325898072270
// MI455X (gfx1250) — matrix operand order repaired and checked by reading, not compiled, not run on hardware
//
#include <hip/hip_runtime.h>
#include <math.h>

// ---------------- types ----------------
typedef __bf16 bf16;
typedef __attribute__((ext_vector_type(16))) __bf16 v16bf;
typedef __attribute__((ext_vector_type(8)))  __bf16 bfx8;
typedef __attribute__((ext_vector_type(8)))  float  v8f;
typedef __attribute__((ext_vector_type(4)))  float  f32x4;

// ---------------- problem dims ----------------
#define T_TOK 8192
#define DIN   512
#define PDIM  1024
#define IDIM  1408
#define NEXP  16
#define KTOP  2
#define ISH   2816   // I * S
#define ADIM  29
#define RSCALE 2.5f

// ---------------- GEMM tiling ----------------
#define BM 128
#define BN 128
#define BK 32
#define BKP 40       // padded LDS row (bank-conflict avoidance, keeps 16B align)
#define NTHR 256

enum { EPI_PROJ = 0, EPI_GATEUP = 1, EPI_STORE = 2, EPI_SILU = 3 };

static __device__ __forceinline__ v16bf cat8(bfx8 lo, bfx8 hi) {
  return __builtin_shufflevector(lo, hi, 0,1,2,3,4,5,6,7,8,9,10,11,12,13,14,15);
}

// Generic tiled bf16 WMMA GEMM: D = epi(A[rows,K] @ B[K,N] + bias)
//  - double-buffered LDS, one barrier per k-tile
//  - bf16 A tiles copied with GLOBAL_LOAD_ASYNC_TO_LDS_B128 (ASYNCcnt fenced)
//  - fp32 A / fp32 B tiles register-prefetched; B packed pairwise (cvt_pk + b32)
//  - gather rows via in_map; expert mode maps blockIdx.x onto ragged segments
template<int EPI, bool AF32>
__global__ __launch_bounds__(NTHR)
void gemm_k(const void* __restrict__ Ap, int lda, int Ktot,
            const float* __restrict__ B0g, const float* __restrict__ B1g,
            int ldb, int Ncols,
            const float* __restrict__ bias0g, const float* __restrict__ bias1g,
            float* __restrict__ outF, bf16* __restrict__ outH, int ldo,
            int Mrows,
            const int* __restrict__ in_map,
            const int* __restrict__ seg_off, long strideB, int strideBias)
{
  constexpr bool GU = (EPI == EPI_GATEUP);

  __shared__ __align__(16) bf16 As [2][BM][BKP];
  __shared__ __align__(16) bf16 Bs0[2][BN][BKP];
  __shared__ __align__(16) bf16 Bs1[GU ? 2 : 1][BN][BKP];

  const float* B0 = B0g;
  const float* B1 = B1g;
  const float* bias0 = bias0g;
  const float* bias1 = bias1g;

  int row0, nrows;
  if (seg_off) {
    int bid = blockIdx.x;
    int e = -1, tile = 0, segN = 0, base = 0, acc = 0;
    for (int i = 0; i < NEXP; ++i) {
      int s0 = seg_off[i], s1 = seg_off[i + 1];
      int nt = (s1 - s0 + BM - 1) / BM;
      if (bid < acc + nt) { e = i; tile = bid - acc; base = s0; segN = s1 - s0; break; }
      acc += nt;
    }
    if (e < 0) return;
    row0  = base + tile * BM;
    nrows = segN - tile * BM; if (nrows > BM) nrows = BM;
    B0 += (long)e * strideB;
    if (B1)    B1    += (long)e * strideB;
    bias0 += (long)e * strideBias;
    if (bias1) bias1 += (long)e * strideBias;
  } else {
    row0  = blockIdx.x * BM;
    nrows = Mrows - row0; if (nrows > BM) nrows = BM;
    if (nrows <= 0) return;
  }
  const int n0 = blockIdx.y * BN;

  const int tid  = threadIdx.x;
  const int lane = tid & 31;
  const int wv   = tid >> 5;
  const int wm   = wv & 3;      // 4 waves over M (32 rows each)
  const int wn   = wv >> 2;     // 2 waves over N (64 cols each)
  const int l15  = lane & 15;
  const int sel  = lane >> 4;   // half-wave selector for k-chunks

  // ---------- hoisted B fill descriptors (clamped -> divergence-free) ----------
  // Each thread owns 4 columns x 2 adjacent-k pairs x 2 halves of the k-tile:
  // pairs (k, k+1) pack into one 32-bit LDS store (v_cvt_pk_bf16_f32 + ds_store_b32).
  const int  b_nc  = (tid & 31) << 2;   // column chunk within tile (0..124)
  const int  b_k2  = (tid >> 5) << 1;   // k-pair base (0,2,...,14); +16 on 2nd iter
  const int  gn    = n0 + b_nc;
  const bool bvec  = (gn + 3 < Ncols) && ((ldb & 3) == 0);
  const int  bcl0  = (gn     < Ncols) ? gn     : Ncols - 1;
  const int  bcl1  = (gn + 1 < Ncols) ? gn + 1 : Ncols - 1;
  const int  bcl2  = (gn + 2 < Ncols) ? gn + 2 : Ncols - 1;
  const int  bcl3  = (gn + 3 < Ncols) ? gn + 3 : Ncols - 1;

  // ---------- hoisted A fill descriptors ----------
  int a_kc = 0, a_kc8 = 0;
  const float* a_ptr32[4] = {nullptr, nullptr, nullptr, nullptr};
  const bf16*  a_gb[2]    = {nullptr, nullptr};
  unsigned     a_lds[2][2] = {};
  if constexpr (AF32) {
    a_kc = (tid & 7) << 2;
    const float* Af = (const float*)Ap;
#pragma unroll
    for (int it = 0; it < 4; ++it) {
      int m   = (tid >> 3) + (it << 5);
      int rcl = (m < nrows) ? m : nrows - 1;      // clamp: OOB rows discarded at store
      int r   = row0 + rcl;
      int ar  = in_map ? in_map[r] : r;
      a_ptr32[it] = Af + (long)ar * lda + a_kc;
    }
  } else {
    a_kc8 = (tid & 3) << 3;
    const bf16* Ah = (const bf16*)Ap;
#pragma unroll
    for (int it = 0; it < 2; ++it) {
      int m   = (tid >> 2) + (it << 6);
      int rcl = (m < nrows) ? m : nrows - 1;
      int r   = row0 + rcl;
      int ar  = in_map ? in_map[r] : r;
      a_gb[it]     = Ah + (long)ar * lda + a_kc8;
      a_lds[it][0] = (unsigned)(size_t)&As[0][m][a_kc8];  // low 32b of generic = LDS addr
      a_lds[it][1] = (unsigned)(size_t)&As[1][m][a_kc8];
    }
  }

  // ---------- fill helpers ----------
  // regs layout: r[2*it+0] = row k, r[2*it+1] = row k+1, it selects k-tile half.
  auto loadB = [&](int k0, f32x4* r0, f32x4* r1) {
#pragma unroll
    for (int it = 0; it < 2; ++it) {
      long kk = k0 + b_k2 + (it << 4);
      const float* bp0 = B0 + kk * ldb;
      const float* bp1 = B0 + (kk + 1) * ldb;
      if (bvec) {
        r0[2*it]   = *(const f32x4*)(bp0 + gn);
        r0[2*it+1] = *(const f32x4*)(bp1 + gn);
      } else {
        r0[2*it][0] = bp0[bcl0]; r0[2*it][1] = bp0[bcl1]; r0[2*it][2] = bp0[bcl2]; r0[2*it][3] = bp0[bcl3];
        r0[2*it+1][0] = bp1[bcl0]; r0[2*it+1][1] = bp1[bcl1]; r0[2*it+1][2] = bp1[bcl2]; r0[2*it+1][3] = bp1[bcl3];
      }
      if constexpr (GU) {
        const float* bq0 = B1 + kk * ldb;
        const float* bq1 = B1 + (kk + 1) * ldb;
        if (bvec) {
          r1[2*it]   = *(const f32x4*)(bq0 + gn);
          r1[2*it+1] = *(const f32x4*)(bq1 + gn);
        } else {
          r1[2*it][0] = bq0[bcl0]; r1[2*it][1] = bq0[bcl1]; r1[2*it][2] = bq0[bcl2]; r1[2*it][3] = bq0[bcl3];
          r1[2*it+1][0] = bq1[bcl0]; r1[2*it+1][1] = bq1[bcl1]; r1[2*it+1][2] = bq1[bcl2]; r1[2*it+1][3] = bq1[bcl3];
        }
      }
    }
  };
  auto storeB = [&](int buf, const f32x4* r0, const f32x4* r1) {
#pragma unroll
    for (int it = 0; it < 2; ++it) {
      int kk = b_k2 + (it << 4);
#pragma unroll
      for (int j = 0; j < 4; ++j) {
        union { bf16 h[2]; unsigned u; } pv;
        pv.h[0] = (bf16)r0[2*it][j];
        pv.h[1] = (bf16)r0[2*it+1][j];
        *(unsigned*)&Bs0[buf][b_nc + j][kk] = pv.u;   // 4B-aligned (kk even)
      }
      if constexpr (GU) {
#pragma unroll
        for (int j = 0; j < 4; ++j) {
          union { bf16 h[2]; unsigned u; } pv;
          pv.h[0] = (bf16)r1[2*it][j];
          pv.h[1] = (bf16)r1[2*it+1][j];
          *(unsigned*)&Bs1[buf][b_nc + j][kk] = pv.u;
        }
      }
    }
  };
  auto loadA32 = [&](int k0, f32x4* ra) {
#pragma unroll
    for (int it = 0; it < 4; ++it) ra[it] = *(const f32x4*)(a_ptr32[it] + k0);
  };
  auto storeA32 = [&](int buf, const f32x4* ra) {
#pragma unroll
    for (int it = 0; it < 4; ++it) {
      int m = (tid >> 3) + (it << 5);
      union { bf16 h[4]; uint2 u; } cv;
#pragma unroll
      for (int j = 0; j < 4; ++j) cv.h[j] = (bf16)ra[it][j];
      *(uint2*)&As[buf][m][a_kc] = cv.u;
    }
  };
  auto issueA = [&](int k0, int buf) {   // async DMA: global -> LDS, ASYNCcnt
#pragma unroll
    for (int it = 0; it < 2; ++it) {
      asm volatile("global_load_async_to_lds_b128 %0, %1, off"
                   :: "v"(a_lds[it][buf]),
                      "v"((unsigned long long)(a_gb[it] + k0))
                   : "memory");
    }
  };

  // ---------- accumulators ----------
  v8f acc0[2][4];
  v8f acc1[2][4];
  const v8f vzero = {0.f,0.f,0.f,0.f,0.f,0.f,0.f,0.f};
#pragma unroll
  for (int i = 0; i < 2; ++i)
#pragma unroll
    for (int j = 0; j < 4; ++j) { acc0[i][j] = vzero; if constexpr (GU) acc1[i][j] = vzero; }

  // ---------- software-pipelined main loop (double-buffered LDS) ----------
  f32x4 rB0[4], rB1[4], rA[4];
  f32x4 nB0[4], nB1[4], nA[4];

  loadB(0, rB0, rB1);
  if constexpr (AF32) loadA32(0, rA);
  storeB(0, rB0, rB1);
  if constexpr (AF32) storeA32(0, rA); else issueA(0, 0);

  const int nk = Ktot / BK;
  int cur = 0;
  for (int kt = 0; kt < nk; ++kt) {
    const int  k0n  = (kt + 1) * BK;
    const bool more = (kt + 1 < nk);
    if (more) {
      loadB(k0n, nB0, nB1);
      if constexpr (AF32) loadA32(k0n, nA);
      if ((kt + 2 < nk) && (n0 + BN <= Ncols))
        __builtin_prefetch(B0 + (long)((kt + 2) * BK + b_k2) * ldb + gn, 0, 1);
    }
    if constexpr (!AF32) asm volatile("s_wait_asynccnt 0x0" ::: "memory");
    __syncthreads();

    // ---- compute on buffer `cur` ----
    v16bf afr[2], bfr0[4], bfr1[4];
#pragma unroll
    for (int mi = 0; mi < 2; ++mi) {
      int m = wm * 32 + mi * 16 + l15;
      bfx8 lo = *(const bfx8*)&As[cur][m][sel * 8];       // K 0..7  (or 8..15)
      bfx8 hi = *(const bfx8*)&As[cur][m][16 + sel * 8];  // K 16..23 (or 24..31)
      afr[mi] = cat8(lo, hi);
    }
#pragma unroll
    for (int ni = 0; ni < 4; ++ni) {
      int n = wn * 64 + ni * 16 + l15;
      {
        bfx8 lo = *(const bfx8*)&Bs0[cur][n][sel * 8];
        bfx8 hi = *(const bfx8*)&Bs0[cur][n][16 + sel * 8];
        bfr0[ni] = cat8(lo, hi);
      }
      if constexpr (GU) {
        bfx8 lo = *(const bfx8*)&Bs1[cur][n][sel * 8];
        bfx8 hi = *(const bfx8*)&Bs1[cur][n][16 + sel * 8];
        bfr1[ni] = cat8(lo, hi);
      }
    }
#pragma unroll
    for (int mi = 0; mi < 2; ++mi)
#pragma unroll
      for (int ni = 0; ni < 4; ++ni) {
        acc0[mi][ni] = __builtin_amdgcn_wmma_f32_16x16x32_bf16(
            false, afr[mi], false, bfr0[ni], (short)0, acc0[mi][ni], false, false);
        if constexpr (GU)
          acc1[mi][ni] = __builtin_amdgcn_wmma_f32_16x16x32_bf16(
              false, afr[mi], false, bfr1[ni], (short)0, acc1[mi][ni], false, false);
      }

    if (more) {   // fill the alternate buffer (no barrier needed: disjoint from `cur`)
      storeB(cur ^ 1, nB0, nB1);
      if constexpr (AF32) storeA32(cur ^ 1, nA); else issueA(k0n, cur ^ 1);
      cur ^= 1;
    }
  }

  // ---------- epilogue ----------
#pragma unroll
  for (int mi = 0; mi < 2; ++mi)
#pragma unroll
    for (int ni = 0; ni < 4; ++ni) {
      int c = n0 + wn * 64 + ni * 16 + l15;
      if (c >= Ncols) continue;
#pragma unroll
      for (int i = 0; i < 8; ++i) {
        int mloc = wm * 32 + mi * 16 + i + 8 * sel;
        if (mloc >= nrows) continue;
        long r = row0 + mloc;
        float v = acc0[mi][ni][i];
        if constexpr (EPI == EPI_PROJ) {
          v += bias0[c];
          outF[r * ldo + c] = v;
          outH[r * ldo + c] = (bf16)v;
        } else if constexpr (EPI == EPI_GATEUP) {
          float g = v + bias0[c];
          float u = acc1[mi][ni][i] + bias1[c];
          float h = (g / (1.f + __expf(-g))) * u;
          outH[r * ldo + c] = (bf16)h;
        } else if constexpr (EPI == EPI_STORE) {
          outF[r * ldo + c] = v + bias0[c];
        } else { // EPI_SILU
          float g = v + bias0[c];
          outH[r * ldo + c] = (bf16)(g / (1.f + __expf(-g)));
        }
      }
    }
}

// ---------------- router: one wave per token ----------------
__global__ __launch_bounds__(256)
void router_k(const float* __restrict__ xp, const float* __restrict__ Wr,
              const float* __restrict__ br, const float* __restrict__ bcorr,
              int* __restrict__ topk_idx, float* __restrict__ topk_w,
              int* __restrict__ counts)
{
  int wv = threadIdx.x >> 5, lane = threadIdx.x & 31;
  int t = blockIdx.x * 8 + wv;
  if (t >= T_TOK) return;

  float part[NEXP];
#pragma unroll
  for (int e = 0; e < NEXP; ++e) part[e] = 0.f;

  const float* xr = xp + (long)t * PDIM;
  for (int p = lane; p < PDIM; p += 32) {
    float xv = xr[p];
    const float* wr = Wr + p * NEXP;
#pragma unroll
    for (int e = 0; e < NEXP; ++e) part[e] += xv * wr[e];
  }
#pragma unroll
  for (int e = 0; e < NEXP; ++e)
#pragma unroll
    for (int off = 16; off; off >>= 1) part[e] += __shfl_xor(part[e], off, 32);

  if (lane != 0) return;

  float scores[NEXP], sfc[NEXP];
#pragma unroll
  for (int e = 0; e < NEXP; ++e) {
    float l = part[e] + br[e];
    scores[e] = 1.f / (1.f + __expf(-l));
    sfc[e] = scores[e] + bcorr[e];
  }
  // group scores: top-2 sum within each group of 4
  float gs[4];
#pragma unroll
  for (int g = 0; g < 4; ++g) {
    float a = sfc[4*g], b = sfc[4*g+1], c = sfc[4*g+2], d = sfc[4*g+3];
    float hi_ab = fmaxf(a, b), hi_cd = fmaxf(c, d);
    float lo_ab = fminf(a, b), lo_cd = fminf(c, d);
    float top = fmaxf(hi_ab, hi_cd);
    float second = fmaxf(fminf(hi_ab, hi_cd), (hi_ab >= hi_cd) ? lo_ab : lo_cd);
    gs[g] = top + second;
  }
  // top-2 groups (ties -> lowest index, like jax.top_k)
  int g1 = 0; float bg1 = -1e30f;
#pragma unroll
  for (int g = 0; g < 4; ++g) if (gs[g] > bg1) { bg1 = gs[g]; g1 = g; }
  int g2 = 0; float bg2 = -1e30f;
#pragma unroll
  for (int g = 0; g < 4; ++g) if (g != g1 && gs[g] > bg2) { bg2 = gs[g]; g2 = g; }
  // mask and pick top-2 experts
  float mk[NEXP];
#pragma unroll
  for (int e = 0; e < NEXP; ++e) {
    int g = e >> 2;
    mk[e] = (g == g1 || g == g2) ? sfc[e] : 0.0f;
  }
  int i1 = 0; float v1 = -1e30f, s1 = 0.f;
#pragma unroll
  for (int e = 0; e < NEXP; ++e) if (mk[e] > v1) { v1 = mk[e]; i1 = e; s1 = scores[e]; }
  int i2 = 0; float v2 = -1e30f, s2 = 0.f;
#pragma unroll
  for (int e = 0; e < NEXP; ++e) if (e != i1 && mk[e] > v2) { v2 = mk[e]; i2 = e; s2 = scores[e]; }

  float s = s1 + s2 + 1e-20f;
  float w1 = s1 / s * RSCALE;
  float w2 = s2 / s * RSCALE;

  topk_idx[2*t]   = i1; topk_w[2*t]   = w1;
  topk_idx[2*t+1] = i2; topk_w[2*t+1] = w2;
  atomicAdd(&counts[i1], 1);
  atomicAdd(&counts[i2], 1);
}

__global__ void zero_k(int* counts) {
  if (threadIdx.x < NEXP) counts[threadIdx.x] = 0;
}

__global__ void scan_k(const int* __restrict__ counts, int* __restrict__ offsets,
                       int* __restrict__ cursor) {
  if (threadIdx.x == 0) {
    int a = 0;
    for (int e = 0; e < NEXP; ++e) { offsets[e] = a; cursor[e] = a; a += counts[e]; }
    offsets[NEXP] = a;
  }
}

__global__ __launch_bounds__(256)
void scatter_k(const int* __restrict__ topk_idx, int* __restrict__ cursor,
               int* __restrict__ assign_token, int* __restrict__ assign_pos) {
  int t = blockIdx.x * blockDim.x + threadIdx.x;
  if (t >= T_TOK) return;
#pragma unroll
  for (int k = 0; k < KTOP; ++k) {
    int e = topk_idx[2*t + k];
    int pos = atomicAdd(&cursor[e], 1);
    assign_token[pos] = t;
    assign_pos[2*t + k] = pos;
  }
}

// combined[t][c] += w0 * eo[pos0][c] + w1 * eo[pos1][c]   (deterministic)
__global__ __launch_bounds__(256)
void combine_k(float* __restrict__ combined, const float* __restrict__ eo,
               const float* __restrict__ topk_w, const int* __restrict__ assign_pos) {
  long idx = (long)blockIdx.x * 256 + threadIdx.x;
  long t = idx >> 10;        // / PDIM
  int  c = (int)(idx & (PDIM - 1));
  float v = combined[idx];
  v += topk_w[2*t]     * eo[(long)assign_pos[2*t]     * PDIM + c];
  v += topk_w[2*t + 1] * eo[(long)assign_pos[2*t + 1] * PDIM + c];
  combined[idx] = v;
}

// ---------------- host launcher ----------------
extern "C" void kernel_launch(void* const* d_in, const int* in_sizes, int n_in,
                              void* d_out, int out_size, void* d_ws, size_t ws_size,
                              hipStream_t stream) {
  const float* x     = (const float*)d_in[0];
  const float* Wp    = (const float*)d_in[1];
  const float* bp    = (const float*)d_in[2];
  const float* Wr    = (const float*)d_in[3];
  const float* br    = (const float*)d_in[4];
  const float* bcorr = (const float*)d_in[5];
  const float* Wg    = (const float*)d_in[6];
  const float* bg    = (const float*)d_in[7];
  const float* Wu    = (const float*)d_in[8];
  const float* bu    = (const float*)d_in[9];
  const float* Wd    = (const float*)d_in[10];
  const float* bd    = (const float*)d_in[11];
  const float* Wsg   = (const float*)d_in[12];
  const float* bsg   = (const float*)d_in[13];
  const float* Wsu   = (const float*)d_in[14];
  const float* bsu   = (const float*)d_in[15];
  const float* Wsd   = (const float*)d_in[16];
  const float* bsd   = (const float*)d_in[17];
  const float* W1    = (const float*)d_in[18];
  const float* b1    = (const float*)d_in[19];
  const float* W2    = (const float*)d_in[20];
  const float* b2    = (const float*)d_in[21];

  char* wsp = (char*)d_ws;
  auto alloc = [&](size_t bytes) -> char* {
    char* p = wsp;
    wsp += (bytes + 255) & ~(size_t)255;
    return p;
  };
  float* xpF      = (float*)alloc((size_t)T_TOK * PDIM * 4);
  bf16*  xpH      = (bf16*) alloc((size_t)T_TOK * PDIM * 2);
  bf16*  hsH      = (bf16*) alloc((size_t)T_TOK * ISH  * 2);
  bf16*  hallH    = (bf16*) alloc((size_t)T_TOK * KTOP * IDIM * 2);
  float* eo       = (float*)alloc((size_t)T_TOK * KTOP * PDIM * 4);
  float* combined = (float*)alloc((size_t)T_TOK * PDIM * 4);
  bf16*  h1H      = (bf16*) alloc((size_t)T_TOK * IDIM * 2);
  int*   topk_idx = (int*)  alloc((size_t)T_TOK * KTOP * 4);
  float* topk_w   = (float*)alloc((size_t)T_TOK * KTOP * 4);
  int*   counts   = (int*)  alloc(64);
  int*   offsets  = (int*)  alloc(128);
  int*   cursor   = (int*)  alloc(64);
  int*   a_token  = (int*)  alloc((size_t)T_TOK * KTOP * 4);
  int*   a_pos    = (int*)  alloc((size_t)T_TOK * KTOP * 4);

  const dim3 blk(NTHR);
  const int MT = T_TOK / BM;                        // 64 token tiles
  const int EXP_SLOTS = (T_TOK * KTOP) / BM + NEXP; // 144 worst-case (expert,tile) slots

  // 1) xp = x @ Wp + bp   (fp32 A converted on the fly)
  gemm_k<EPI_PROJ, true><<<dim3(MT, PDIM / BN), blk, 0, stream>>>(
      x, DIN, DIN, Wp, nullptr, PDIM, PDIM, bp, nullptr,
      xpF, xpH, PDIM, T_TOK, nullptr, nullptr, 0, 0);

  // 2) router -> top-2 experts + weights per token, expert counts
  zero_k<<<1, 32, 0, stream>>>(counts);
  router_k<<<dim3(T_TOK / 8), blk, 0, stream>>>(xpF, Wr, br, bcorr,
                                                topk_idx, topk_w, counts);
  scan_k<<<1, 32, 0, stream>>>(counts, offsets, cursor);
  scatter_k<<<dim3(T_TOK / 256), blk, 0, stream>>>(topk_idx, cursor, a_token, a_pos);

  // 3) shared expert: hs = silu(xp@Wsg+bsg)*(xp@Wsu+bsu); combined = hs@Wsd+bsd
  gemm_k<EPI_GATEUP, false><<<dim3(MT, ISH / BN), blk, 0, stream>>>(
      xpH, PDIM, PDIM, Wsg, Wsu, ISH, ISH, bsg, bsu,
      nullptr, hsH, ISH, T_TOK, nullptr, nullptr, 0, 0);
  gemm_k<EPI_STORE, false><<<dim3(MT, PDIM / BN), blk, 0, stream>>>(
      hsH, ISH, ISH, Wsd, nullptr, PDIM, PDIM, bsd, nullptr,
      combined, nullptr, PDIM, T_TOK, nullptr, nullptr, 0, 0);

  // 4) routed experts: gathered gate/up then down per assignment
  gemm_k<EPI_GATEUP, false><<<dim3(EXP_SLOTS, IDIM / BN), blk, 0, stream>>>(
      xpH, PDIM, PDIM, Wg, Wu, IDIM, IDIM, bg, bu,
      nullptr, hallH, IDIM, 0, a_token, offsets, (long)PDIM * IDIM, IDIM);
  gemm_k<EPI_STORE, false><<<dim3(EXP_SLOTS, PDIM / BN), blk, 0, stream>>>(
      hallH, IDIM, IDIM, Wd, nullptr, PDIM, PDIM, bd, nullptr,
      eo, nullptr, PDIM, 0, nullptr, offsets, (long)IDIM * PDIM, PDIM);

  // 5) deterministic weighted combine into `combined`
  combine_k<<<dim3((T_TOK * PDIM) / 256), blk, 0, stream>>>(combined, eo, topk_w, a_pos);

  // 6) output MLP: h1 = silu(combined@W1+b1); out = h1@W2+b2
  gemm_k<EPI_SILU, true><<<dim3(MT, IDIM / BN), blk, 0, stream>>>(
      combined, PDIM, PDIM, W1, nullptr, IDIM, IDIM, b1, nullptr,
      nullptr, h1H, IDIM, T_TOK, nullptr, nullptr, 0, 0);
  gemm_k<EPI_STORE, false><<<dim3(MT, 1), blk, 0, stream>>>(
      h1H, IDIM, IDIM, W2, nullptr, ADIM, ADIM, b2, nullptr,
      (float*)d_out, nullptr, ADIM, T_TOK, nullptr, nullptr, 0, 0);
}
